// SparseConv1dNeq_30451318128899
// MI455X (gfx1250) — matrix operand order repaired and checked by reading, not compiled, not run on hardware
//
#include <hip/hip_runtime.h>

typedef _Float16 half_t;
typedef __attribute__((ext_vector_type(16))) _Float16 v16h;
typedef __attribute__((ext_vector_type(8)))  _Float16 v8h;
typedef __attribute__((ext_vector_type(2)))  _Float16 v2h;
typedef __attribute__((ext_vector_type(8)))  float    v8f;

#define IN_CH 64
#define OUT_CH 64
#define KW 3
#define LLEN 1024
#define TILE 128
#define HALO_COLS (TILE + 2)
#define PITCH 72   // halves per LDS column row; PITCH*2=144 B (16B-aligned strides)

// ---------------------------------------------------------------------------
// Setup: pack (weight * mask) into f16 WMMA A-matrix fragments in d_ws.
// GEMM K ordering: K = k*64 + c, split into 6 chunks of 32 (chunk q: k=q/2,
// c base = 32*(q&1)). A-layout per CDNA5 ISA (16-bit A 16x32):
//   lane 0-15 : M=lane,   elements 0..7 -> K 0..7,  8..15 -> K 16..23
//   lane 16-31: M=lane-16, elements 0..7 -> K 8..15, 8..15 -> K 24..31
// Fragment f = t*6+q (t = o-tile 0..3) stored as ws[f*512 + lane*16 + e].
// ---------------------------------------------------------------------------
__global__ void pack_weights_kernel(const float* __restrict__ w,
                                    const float* __restrict__ m,
                                    half_t* __restrict__ wsA) {
  int idx = blockIdx.x * 256 + threadIdx.x;     // 24*32*16 = 12288 total
  if (idx >= 24 * 32 * 16) return;
  int f = idx >> 9;
  int l = (idx >> 4) & 31;
  int e = idx & 15;
  int t = f / 6, q = f % 6;
  int k = q >> 1, c0 = (q & 1) * 32;
  int o = t * 16 + (l & 15);
  int kl = (e < 8) ? ((l < 16) ? e : e + 8)
                   : ((l < 16) ? e + 8 : e + 16);
  int c = c0 + kl;
  float wv = w[(o * IN_CH + c) * KW + k] * m[(o * IN_CH + c) * KW + k];
  wsA[idx] = (half_t)wv;
}

// ---------------------------------------------------------------------------
// Main kernel: one block = (sample n) x (128 output columns).
// ---------------------------------------------------------------------------
__global__ __launch_bounds__(256)
void sparse_conv1d_wmma_kernel(const float* __restrict__ x,
                               const half_t* __restrict__ wsA,
                               float* __restrict__ out) {
  __shared__ half_t ldsX[HALO_COLS * PITCH];   // [col][chan], f16, 18720 B

  const int tid = threadIdx.x;
  const int n   = blockIdx.y;
  const int l0  = blockIdx.x * TILE;

  // ---- stage + input fake-quant into LDS (transposed) ----
  const float* xb = x + (size_t)n * IN_CH * LLEN;
  for (int i = tid; i < 32 * HALO_COLS; i += 256) {
    int cp  = i / HALO_COLS;          // channel pair 0..31
    int col = i % HALO_COLS;          // 0..129 (halo for padding=1)
    int gl  = l0 + col - 1;
    float a = 0.f, b = 0.f;
    if (gl >= 0 && gl < LLEN) {
      a = __builtin_nontemporal_load(&xb[(2 * cp    ) * LLEN + gl]);
      b = __builtin_nontemporal_load(&xb[(2 * cp + 1) * LLEN + gl]);
    }
    a = fminf(fmaxf(rintf(a * 16.f), -128.f), 127.f) * 0.0625f;
    b = fminf(fmaxf(rintf(b * 16.f), -128.f), 127.f) * 0.0625f;
    v2h p; p.x = (half_t)a; p.y = (half_t)b;
    *(v2h*)&ldsX[col * PITCH + 2 * cp] = p;
  }
  __syncthreads();

  // ---- per-wave: one 16-column tile, all 4 o-tiles ----
  const int wv      = tid >> 5;
  const int lane    = tid & 31;
  const int colTile = wv;                       // 0..7
  v8f acc[4] = {};

  #pragma unroll
  for (int q = 0; q < 6; ++q) {
    const int k  = q >> 1;
    const int c0 = (q & 1) * 32;
    // B-layout (16-bit B, K=32): lane 0-15 -> N=lane, K=e (e<8), 16+e-8 (e>=8); lane 16-31 -> K=8+e, 24+e-8
    const int colb = colTile * 16 + (lane & 15) + k;   // LDS col incl. halo shift
    const half_t* bp = &ldsX[colb * PITCH + c0 + ((lane & 16) ? 8 : 0)];
    v8h blo = *(const v8h*)bp;         // 16B-aligned ds_load_b128
    v8h bhi = *(const v8h*)(bp + 16);
    v16h bm;
    #pragma unroll
    for (int i = 0; i < 8; ++i) { bm[i] = blo[i]; bm[i + 8] = bhi[i]; }

    #pragma unroll
    for (int t = 0; t < 4; ++t) {
      v16h am = *(const v16h*)(wsA + (((t * 6 + q) << 9) + lane * 16));
      acc[t] = __builtin_amdgcn_wmma_f32_16x16x32_f16(
          false, am, false, bm, (short)0, acc[t], false, false);
    }
  }

  // ---- output fake-quant + non-temporal store ----
  const int colg = l0 + colTile * 16 + (lane & 15);
  float* op = out + (size_t)n * OUT_CH * LLEN + colg;
  const int ob = (lane & 16) ? 8 : 0;   // D layout: VGPR r -> M=r (+8 for hi lanes)
  #pragma unroll
  for (int t = 0; t < 4; ++t) {
    #pragma unroll
    for (int r = 0; r < 8; ++r) {
      int o = t * 16 + ob + r;
      float v = acc[t][r];
      v = fminf(fmaxf(rintf(v * 8.f), -128.f), 127.f) * 0.125f;
      __builtin_nontemporal_store(v, &op[(size_t)o * LLEN]);
    }
  }
}

// ---------------------------------------------------------------------------
extern "C" void kernel_launch(void* const* d_in, const int* in_sizes, int n_in,
                              void* d_out, int out_size, void* d_ws, size_t ws_size,
                              hipStream_t stream) {
  const float* x  = (const float*)d_in[0];   // [N, 64, 1024] f32
  const float* w  = (const float*)d_in[1];   // [64, 64, 3]   f32
  const float* mk = (const float*)d_in[2];   // [64, 64, 3]   f32
  float* out = (float*)d_out;
  half_t* wsA = (half_t*)d_ws;               // 24 KB of packed A fragments

  const int N = in_sizes[0] / (IN_CH * LLEN);

  pack_weights_kernel<<<48, 256, 0, stream>>>(w, mk, wsA);
  dim3 grid(LLEN / TILE, N);
  sparse_conv1d_wmma_kernel<<<grid, 256, 0, stream>>>(x, wsA, out);
}
